// VisionMambaLite_56968446214313
// MI455X (gfx1250) — matrix operand order repaired and checked by reading, not compiled, not run on hardware
//
#include <hip/hip_runtime.h>
#include <hip/hip_bf16.h>

// ---------------------------------------------------------------------------
// VisionMambaLite forward for gfx1250 (MI455X), bf16 WMMA + f32 accumulate.
// ---------------------------------------------------------------------------

#define DEPTH 6
#define DIM   512
#define DI    768
#define BSZ   64
#define HW    14
#define NP    (HW*HW)        // 196
#define NTOK  (BSZ*NP)       // 12544
#define NSEQ  (BSZ*HW)       // 896
#define D2    (2*DI)         // 1536
#define FK    (2*DIM)        // 1024

typedef __attribute__((ext_vector_type(16))) __bf16       v16bf;
typedef __attribute__((ext_vector_type(8)))  float        v8f;
typedef __attribute__((ext_vector_type(4)))  unsigned int u32x4;

union Frag { v16bf v; u32x4 q[2]; };

__device__ __forceinline__ float siluf(float x)     { return x / (1.f + __expf(-x)); }
__device__ __forceinline__ float softplusf(float x) { return (x > 20.f) ? x : log1pf(__expf(x)); }

// ---------------------------------------------------------------------------
// GEMM: C(M x N) = A(M x K, bf16 row-major) * Bt(N x K, bf16 row-major)^T
// Block = 128M x 128N, 8 waves (2Mx4N), each wave computes 64M x 32N with
// v_wmma_f32_16x16x32_bf16: 8 accumulators + 4 A-frags + 2 B-frags (~130
// VGPRs) so the k-loop stays fused. Fragments load straight from global
// (all operands L2-resident on MI455X's 192MB L2; no LDS staging needed).
// epi bits: 1=+bias[col]  2=softplus  4=+pos_embed[(row%196)+1]  8=store bf16
// ---------------------------------------------------------------------------
__global__ __launch_bounds__(256, 1)
void gemm_bf16_wmma(const __bf16* __restrict__ A, int lda,
                    const __bf16* __restrict__ Bt, int ldb,
                    const float* __restrict__ bias,
                    const float* __restrict__ pos,
                    float* __restrict__ Cf, __bf16* __restrict__ Ch,
                    int ldc, int cOff, int K, int epi)
{
    const int lane = threadIdx.x & 31;
    const int wave = threadIdx.x >> 5;
    const int half = lane >> 4;      // 0: lanes 0-15, 1: lanes 16-31
    const int r    = lane & 15;
    const int mBase = blockIdx.y * 128 + (wave >> 2) * 64;
    const int nBase = blockIdx.x * 128 + (wave & 3) * 32;

    v8f acc[4][2];
    const v8f vz = {0.f,0.f,0.f,0.f,0.f,0.f,0.f,0.f};
#pragma unroll
    for (int mt = 0; mt < 4; ++mt)
#pragma unroll
        for (int nt = 0; nt < 2; ++nt)
            acc[mt][nt] = vz;

    const __bf16* aRow = A  + (size_t)(mBase + r) * lda + half*8;
    const __bf16* bRow = Bt + (size_t)(nBase + r) * ldb + half*8;

    for (int k0 = 0; k0 < K; k0 += 32) {
        Frag a[4];
#pragma unroll
        for (int mt = 0; mt < 4; ++mt) {
            // 16-bit A 16x32 layout: lanes 0-15 hold M=r, K = k0+{0..7, 16..23};
            // lanes 16-31 hold M=r, K = k0+{8..15, 24..31}.
            const __bf16* ap = aRow + (size_t)mt*16*lda + k0;
            a[mt].q[0] = *(const u32x4*)ap;
            a[mt].q[1] = *(const u32x4*)(ap + 16);
        }
        Frag b[2];
#pragma unroll
        for (int nt = 0; nt < 2; ++nt) {
            // B 32x16 layout: lanes 0-15 hold N=r, K = k0+{0..7, 16..23};
            // lanes 16-31 hold N=r, K = k0+{8..15, 24..31}.
            const __bf16* bp = bRow + (size_t)nt*16*ldb + k0;
            b[nt].q[0] = *(const u32x4*)bp;
            b[nt].q[1] = *(const u32x4*)(bp + 16);
        }
#pragma unroll
        for (int nt = 0; nt < 2; ++nt)
#pragma unroll
            for (int mt = 0; mt < 4; ++mt)
                acc[mt][nt] = __builtin_amdgcn_wmma_f32_16x16x32_bf16(
                    false, a[mt].v, false, b[nt].v, (short)0, acc[mt][nt], false, false);
    }

    // Epilogue. C/D layout: VGPR j -> row = half*8 + j, col = r (within tile).
#pragma unroll
    for (int nt = 0; nt < 2; ++nt) {
        const int col = nBase + nt*16 + r;
        const float bv = (epi & 1) ? bias[col] : 0.f;
#pragma unroll
        for (int mt = 0; mt < 4; ++mt) {
#pragma unroll
            for (int j = 0; j < 8; ++j) {
                const int row = mBase + mt*16 + half*8 + j;
                float v = acc[mt][nt][j] + bv;
                if (epi & 4) v += pos[((row % NP) + 1) * DIM + col];
                if (epi & 2) v = softplusf(v);
                const size_t o = (size_t)row * ldc + cOff + col;
                if (epi & 8) Ch[o] = (__bf16)v;
                else         Cf[o] = v;
            }
        }
    }
}

// ---------------------------------------------------------------------------
// LayerNorm over rows of width 512 (one wave32 per row, 16 elems/lane).
// addInput: out = x + LN(x)  (cls path).  Res: out = LN(x + Res).
// Output row scatter: orow = (row/grp)*outStride + row%grp + outOff
// (grp=R,outStride=0 -> identity; used to interleave into (B,197,512)).
// ---------------------------------------------------------------------------
__global__ __launch_bounds__(256)
void ln_rows(const float* __restrict__ X, const float* __restrict__ Res,
             const float* __restrict__ w, const float* __restrict__ b,
             float* __restrict__ outF, __bf16* __restrict__ outH,
             int R, int addInput, int grp, int outStride, int outOff)
{
    const int row = blockIdx.x * 8 + (threadIdx.x >> 5);
    if (row >= R) return;
    const int lane = threadIdx.x & 31;
    const float* x = X + (size_t)row * DIM;
    float v[16];
    float s = 0.f;
#pragma unroll
    for (int i = 0; i < 16; ++i) {
        const int c = lane + 32*i;
        float t = x[c];
        if (Res) t += Res[(size_t)row*DIM + c];
        v[i] = t; s += t;
    }
#pragma unroll
    for (int off = 16; off > 0; off >>= 1) s += __shfl_xor(s, off, 32);
    const float mean = s * (1.f/DIM);
    float q = 0.f;
#pragma unroll
    for (int i = 0; i < 16; ++i) { float d0 = v[i]-mean; q += d0*d0; }
#pragma unroll
    for (int off = 16; off > 0; off >>= 1) q += __shfl_xor(q, off, 32);
    const float inv = rsqrtf(q*(1.f/DIM) + 1e-5f);
    const size_t orow = (size_t)(row / grp) * outStride + (row % grp) + outOff;
#pragma unroll
    for (int i = 0; i < 16; ++i) {
        const int c = lane + 32*i;
        float nv = (v[i]-mean)*inv*w[c] + b[c];
        if (addInput) nv += v[i];
        if (outF) outF[orow*DIM + c] = nv;
        if (outH) outH[orow*DIM + c] = (__bf16)nv;
    }
}

// ---------------------------------------------------------------------------
// Depthwise conv1d (width 3, zero pad) + SiLU over sequences of length 14.
// dirV=0: sequences are rows (token = b*196 + i*14 + l)
// dirV=1: sequences are cols (token = b*196 + l*14 + i)
// Reads xi = xz[:, 0:768]; writes xc in f32 and bf16.
// ---------------------------------------------------------------------------
__global__ void conv1d_silu(const float* __restrict__ xz,
                            const float* __restrict__ cw, const float* __restrict__ cb,
                            float* __restrict__ xcF, __bf16* __restrict__ xcH, int dirV)
{
    const int idx = blockIdx.x * blockDim.x + threadIdx.x;
    if (idx >= NSEQ * DI) return;
    const int d = idx % DI;
    const int s = idx / DI;
    const int o = s / HW, ii = s % HW;
    const size_t tok0 = (size_t)o * NP + (dirV ? ii : ii * HW);
    const int step = dirV ? HW : 1;
    float xi[HW];
#pragma unroll
    for (int l = 0; l < HW; ++l) xi[l] = xz[(tok0 + (size_t)l*step)*D2 + d];
    const float w0 = cw[d*3+0], w1 = cw[d*3+1], w2 = cw[d*3+2], bb = cb[d];
#pragma unroll
    for (int l = 0; l < HW; ++l) {
        float u = (l > 0 ? xi[l-1] : 0.f)*w0 + xi[l]*w1 + (l < HW-1 ? xi[l+1] : 0.f)*w2 + bb;
        float sc = siluf(u);
        const size_t t = tok0 + (size_t)l*step;
        xcF[t*DI + d] = sc;
        xcH[t*DI + d] = (__bf16)sc;
    }
}

// ---------------------------------------------------------------------------
// SSM scan:  Ae = exp(A*dt); y = cumsum(xc*dt*Ae)*Ae + xc*D; y *= silu(z)
// z = xz[:, 768:1536]. Writes y as bf16 (feeds the out-projection GEMM).
// ---------------------------------------------------------------------------
__global__ void ssm_scan(const float* __restrict__ xcF, const float* __restrict__ dtF,
                         const float* __restrict__ xz,
                         const float* __restrict__ Ap, const float* __restrict__ Dp,
                         __bf16* __restrict__ yH, int dirV)
{
    const int idx = blockIdx.x * blockDim.x + threadIdx.x;
    if (idx >= NSEQ * DI) return;
    const int d = idx % DI;
    const int s = idx / DI;
    const int o = s / HW, ii = s % HW;
    const size_t tok0 = (size_t)o * NP + (dirV ? ii : ii * HW);
    const int step = dirV ? HW : 1;
    const float a = Ap[d], Dd = Dp[d];
    float acc = 0.f;
    for (int l = 0; l < HW; ++l) {
        const size_t t = tok0 + (size_t)l*step;
        const float xcv = xcF[t*DI + d];
        const float dtv = dtF[t*DI + d];
        const float Ae  = __expf(a * dtv);
        acc = fmaf(xcv * dtv, Ae, acc);
        float y = acc * Ae + xcv * Dd;
        const float z = xz[t*D2 + DI + d];
        y *= siluf(z);
        yH[t*DI + d] = (__bf16)y;
    }
}

// im2col for the 16x16/stride16 patch conv: col[(b*196+p), c*256+i*16+j]
__global__ void im2col_patch(const float* __restrict__ x, __bf16* __restrict__ col)
{
    const int idx = blockIdx.x * blockDim.x + threadIdx.x;
    if (idx >= NTOK * DI) return;
    const int cc = idx % DI;
    const int t  = idx / DI;
    const int b = t / NP, p = t % NP;
    const int ph = p / HW, pw = p % HW;
    const int c = cc >> 8, rem = cc & 255;
    const int i = rem >> 4, j = rem & 15;
    const size_t src = (((size_t)b*3 + c)*224 + (ph*16 + i))*224 + (pw*16 + j);
    col[idx] = (__bf16)x[src];
}

__global__ void cls_init(const float* __restrict__ clsTok, const float* __restrict__ pos,
                         float* __restrict__ cls)
{
    const int idx = blockIdx.x * blockDim.x + threadIdx.x;
    if (idx >= BSZ * DIM) return;
    const int n = idx % DIM;
    cls[idx] = clsTok[n] + pos[n];   // pos row 0 = cls position
}

__global__ void cvt_bf16(const float* __restrict__ s, __bf16* __restrict__ d, int n)
{
    const int i = blockIdx.x * blockDim.x + threadIdx.x;
    if (i < n) d[i] = (__bf16)s[i];
}

// ---------------------------------------------------------------------------
extern "C" void kernel_launch(void* const* d_in, const int* in_sizes, int n_in,
                              void* d_out, int out_size, void* d_ws, size_t ws_size,
                              hipStream_t stream)
{
    (void)in_sizes; (void)n_in; (void)out_size; (void)ws_size;
    const float* x        = (const float*)d_in[0];
    const float* patch_w  = (const float*)d_in[1];
    const float* patch_b  = (const float*)d_in[2];
    const float* cls_tok  = (const float*)d_in[3];
    const float* pos      = (const float*)d_in[4];
    const float* blk_w    = (const float*)d_in[5];
    const float* blk_b    = (const float*)d_in[6];
    const float* m_in_w[2]  = { (const float*)d_in[7],  (const float*)d_in[15] };
    const float* m_out_w[2] = { (const float*)d_in[8],  (const float*)d_in[16] };
    const float* m_cw[2]    = { (const float*)d_in[9],  (const float*)d_in[17] };
    const float* m_cb[2]    = { (const float*)d_in[10], (const float*)d_in[18] };
    const float* m_dtw[2]   = { (const float*)d_in[11], (const float*)d_in[19] };
    const float* m_dtb[2]   = { (const float*)d_in[12], (const float*)d_in[20] };
    const float* m_A[2]     = { (const float*)d_in[13], (const float*)d_in[21] };
    const float* m_D[2]     = { (const float*)d_in[14], (const float*)d_in[22] };
    const float* fus_w    = (const float*)d_in[23];
    const float* fus_b    = (const float*)d_in[24];
    const float* ss_w     = (const float*)d_in[25];
    const float* ss_b     = (const float*)d_in[26];
    const float* fn_w     = (const float*)d_in[27];
    const float* fn_b     = (const float*)d_in[28];
    float* out = (float*)d_out;

    // ---- workspace layout (all 256B aligned) ----
    char* base = (char*)d_ws;
    size_t off = 0;
    auto alloc = [&](size_t bytes) -> char* {
        char* p = base + off;
        off += (bytes + 255) & ~(size_t)255;
        return p;
    };
    float*  pt   = (float*) alloc((size_t)NTOK*DIM*4);
    float*  pn   = (float*) alloc((size_t)NTOK*DIM*4);
    __bf16* pn16 = (__bf16*)alloc((size_t)NTOK*DIM*2);
    float*  xz   = (float*) alloc((size_t)NTOK*D2*4);      // also aliased by col16/yf
    float*  xcF  = (float*) alloc((size_t)NTOK*DI*4);
    __bf16* xcH  = (__bf16*)alloc((size_t)NTOK*DI*2);
    float*  dtF  = (float*) alloc((size_t)NTOK*DI*4);
    __bf16* y16  = (__bf16*)alloc((size_t)NTOK*DI*2);
    __bf16* yc16 = (__bf16*)alloc((size_t)NTOK*FK*2);
    float*  cls  = (float*) alloc((size_t)BSZ*DIM*4);
    __bf16* w_patch16 = (__bf16*)alloc((size_t)DIM*DI*2);
    __bf16* w_in16[2]  = { (__bf16*)alloc((size_t)DEPTH*D2*DIM*2),
                           (__bf16*)alloc((size_t)DEPTH*D2*DIM*2) };
    __bf16* w_dt16[2]  = { (__bf16*)alloc((size_t)DEPTH*DI*DI*2),
                           (__bf16*)alloc((size_t)DEPTH*DI*DI*2) };
    __bf16* w_out16[2] = { (__bf16*)alloc((size_t)DEPTH*DIM*DI*2),
                           (__bf16*)alloc((size_t)DEPTH*DIM*DI*2) };
    __bf16* w_fus16    = (__bf16*)alloc((size_t)DEPTH*DIM*FK*2);
    // aliases (disjoint lifetimes vs xz):
    __bf16* col16 = (__bf16*)xz;   // used only before layer loop
    float*  yf    = (float*)xz;    // used only after both scans of a layer

    auto cvt = [&](const float* s, __bf16* d, int n) {
        cvt_bf16<<<(n + 255)/256, 256, 0, stream>>>(s, d, n);
    };
    auto gemm = [&](const __bf16* A, int lda, const __bf16* Bt, int ldb,
                    const float* bias, const float* posP,
                    float* Cf, __bf16* Ch, int ldc, int cOff,
                    int N, int K, int epi) {
        dim3 grid(N/128, NTOK/128);
        gemm_bf16_wmma<<<grid, 256, 0, stream>>>(A, lda, Bt, ldb, bias, posP,
                                                 Cf, Ch, ldc, cOff, K, epi);
    };

    // ---- weight conversion to bf16 (all weights already (N,K) row-major) ----
    cvt(patch_w, w_patch16, DIM*DI);
    for (int d = 0; d < 2; ++d) {
        cvt(m_in_w[d],  w_in16[d],  DEPTH*D2*DIM);
        cvt(m_dtw[d],   w_dt16[d],  DEPTH*DI*DI);
        cvt(m_out_w[d], w_out16[d], DEPTH*DIM*DI);
    }
    cvt(fus_w, w_fus16, DEPTH*DIM*FK);

    // ---- patch embed + pos + cls ----
    im2col_patch<<<(NTOK*DI + 255)/256, 256, 0, stream>>>(x, col16);
    gemm(col16, DI, w_patch16, DI, patch_b, pos, pt, nullptr, DIM, 0, DIM, DI, 1|4);
    cls_init<<<(BSZ*DIM + 255)/256, 256, 0, stream>>>(cls_tok, pos, cls);

    const int lnGrid = (NTOK + 7)/8;   // 1568
    const int ewGrid = (NSEQ*DI + 255)/256;

    for (int i = 0; i < DEPTH; ++i) {
        // pn = LN(pt)  (f32 + bf16 copy for GEMM A)
        ln_rows<<<lnGrid, 256, 0, stream>>>(pt, nullptr, blk_w + i*DIM, blk_b + i*DIM,
                                            pn, pn16, NTOK, 0, NTOK, 0, 0);
        for (int dir = 0; dir < 2; ++dir) {
            const __bf16* w_in  = w_in16[dir]  + (size_t)i*D2*DIM;
            const __bf16* w_dt  = w_dt16[dir]  + (size_t)i*DI*DI;
            const __bf16* w_out = w_out16[dir] + (size_t)i*DIM*DI;
            // xz = pn @ in_w^T
            gemm(pn16, DIM, w_in, DIM, nullptr, nullptr, xz, nullptr, D2, 0, D2, DIM, 0);
            // xc = silu(conv1d(xi) + cb)
            conv1d_silu<<<ewGrid, 256, 0, stream>>>(xz, m_cw[dir] + (size_t)i*DI*3,
                                                    m_cb[dir] + (size_t)i*DI, xcF, xcH, dir);
            // dt = softplus(xc @ dt_w^T + db)
            gemm(xcH, DI, w_dt, DI, m_dtb[dir] + (size_t)i*DI, nullptr,
                 dtF, nullptr, DI, 0, DI, DI, 1|2);
            // scan + gate
            ssm_scan<<<ewGrid, 256, 0, stream>>>(xcF, dtF, xz,
                                                 m_A[dir] + (size_t)i*DI,
                                                 m_D[dir] + (size_t)i*DI, y16, dir);
            // y @ out_w^T -> bf16 into yc columns [dir*512, dir*512+512)
            gemm(y16, DI, w_out, DI, nullptr, nullptr,
                 nullptr, yc16, FK, dir*DIM, DIM, DI, 8);
        }
        // yf = yc @ fus_w^T + fus_b
        gemm(yc16, FK, w_fus16 + (size_t)i*DIM*FK, FK, fus_b + i*DIM, nullptr,
             yf, nullptr, DIM, 0, DIM, FK, 1);
        // pt = LN(yf + pn)   (ss_norm)
        ln_rows<<<lnGrid, 256, 0, stream>>>(yf, pn, ss_w + i*DIM, ss_b + i*DIM,
                                            pt, nullptr, NTOK, 0, NTOK, 0, 0);
        // cls = cls + LN(cls)  (blk_norm, in place)
        ln_rows<<<8, 256, 0, stream>>>(cls, nullptr, blk_w + i*DIM, blk_b + i*DIM,
                                       cls, nullptr, BSZ, 1, BSZ, 0, 0);
    }

    // ---- final LN, scattered into (B, 197, 512) output ----
    ln_rows<<<8, 256, 0, stream>>>(cls, nullptr, fn_w, fn_b, out, nullptr,
                                   BSZ, 0, /*grp*/1, /*stride*/197, /*off*/0);
    ln_rows<<<lnGrid, 256, 0, stream>>>(pt, nullptr, fn_w, fn_b, out, nullptr,
                                        NTOK, 0, /*grp*/NP, /*stride*/197, /*off*/1);
}
